// ResidualDecBlock_50105088475513
// MI455X (gfx1250) — matrix operand order repaired and checked by reading, not compiled, not run on hardware
//
#include <hip/hip_runtime.h>
#include <hip/hip_bf16.h>

// ---------------------------------------------------------------------------
// Residual DGCNN block for MI455X (gfx1250, wave32, WMMA).
// Pipeline: prep -> knn1(WMMA gram, async double-buffered tiles)
//           -> edgeconv1 (2-pass WMMA + BN stats) -> bn1+relu prep
//           -> knn2 -> edgeconv2 -> proj -> fused epilogue.
// gfx1250 paths used: v_wmma_f32_16x16x32_bf16, global_load_async_to_lds_b128
// (+ s_wait_asynccnt), global_prefetch_b8, ds_add_f32 reductions.
// ---------------------------------------------------------------------------

#define BGRAPHS 8
#define NPER    4096
#define NPTS    (BGRAPHS * NPER)   // 32768
#define KNN     16
#define DIN1    64
#define DOUT    128
#define NEDGE   (NPTS * KNN)       // 524288
#define EPSBN   1e-5f

typedef __attribute__((ext_vector_type(16))) __bf16 v16bf;
typedef __attribute__((ext_vector_type(8)))  float  v8f;
typedef int v4i __attribute__((vector_size(16)));

#define AS1 __attribute__((address_space(1)))
#define AS3 __attribute__((address_space(3)))

#if defined(__has_builtin)
#if __has_builtin(__builtin_amdgcn_global_load_async_to_lds_b128)
#define HAVE_ASYNC_LDS_BUILTIN 1
#endif
#if __has_builtin(__builtin_amdgcn_s_wait_asynccnt)
#define HAVE_WAIT_ASYNC_BUILTIN 1
#endif
#endif

// 16B global -> LDS async copy (ASYNCcnt), one lane = one 16B chunk.
__device__ inline void async_cp16(const void* gsrc, void* lds_dst) {
#if defined(HAVE_ASYNC_LDS_BUILTIN)
  __builtin_amdgcn_global_load_async_to_lds_b128((AS1 v4i*)gsrc, (AS3 v4i*)lds_dst,
                                                 0, 0);
#else
  asm volatile("global_load_async_to_lds_b128 %0, %1, off"
               :: "v"((unsigned)(unsigned long long)lds_dst),
                  "v"((unsigned long long)gsrc)
               : "memory");
#endif
}

template <int N>
__device__ inline void wait_async() {
#if defined(HAVE_WAIT_ASYNC_BUILTIN)
  __builtin_amdgcn_s_wait_asynccnt(N);
#else
  asm volatile("s_wait_asynccnt %0" :: "i"(N) : "memory");
#endif
}

// Copy `BYTES` (multiple of 4096) bytes global -> LDS with 256 threads.
// Issues BYTES/4096 ASYNCcnt ops per wave; uniform control flow (EXEC all 1s).
template <int BYTES>
__device__ inline void async_tile_copy(void* lds_dst, const void* gsrc, int tid) {
  const char* g = (const char*)gsrc;
  char* l = (char*)lds_dst;
#pragma unroll
  for (int off = 0; off < BYTES; off += 256 * 16)
    async_cp16(g + off + tid * 16, l + off + tid * 16);
}

// --- WMMA operand loaders (contiguous k order: k = 16*half + s, A and B alike) ------

// A-matrix 16x32 bf16 from row-major [16][ld] tile.
// lane 0-15: M=lane; VGPR v -> K={2v,2v+1}; +16 upper half.
__device__ inline v16bf lds_A_16x32(const __bf16* base, int ld, int lane) {
  const int m = lane & 15, half = lane >> 4;
  v16bf a;
#pragma unroll
  for (int v = 0; v < 8; ++v) {
    const int k = 2 * v + 16 * half;
    a[2 * v]     = base[m * ld + k];
    a[2 * v + 1] = base[m * ld + k + 1];
  }
  return a;
}

// B-matrix 32x16 where B[k][n] = T[n][k], T row-major [16][ld] (Gram: D = A*T^T).
__device__ inline v16bf lds_BT_16x32(const __bf16* base, int ld, int lane) {
  const int n = lane & 15, half = lane >> 4;
  v16bf b;
#pragma unroll
  for (int v = 0; v < 8; ++v) {
    const int k = 2 * v + 16 * half;
    b[2 * v]     = base[n * ld + k];
    b[2 * v + 1] = base[n * ld + k + 1];
  }
  return b;
}

// B-matrix 32x16 from K-major weight Wt[CD][128] in global (L2-resident).
__device__ inline v16bf gbl_B_Kmajor(const __bf16* __restrict__ Wt, int kk, int col0, int lane) {
  const int n = lane & 15, half = lane >> 4;
  v16bf b;
#pragma unroll
  for (int v = 0; v < 8; ++v) {
    const int k = kk + 2 * v + 16 * half;
    b[2 * v]     = Wt[k * DOUT + col0 + n];
    b[2 * v + 1] = Wt[(k + 1) * DOUT + col0 + n];
  }
  return b;
}

// --- small helpers ----------------------------------------------------------

__global__ void zero_kernel(float* __restrict__ p, int n) {
  const int t = blockIdx.x * 256 + threadIdx.x;
  if (t < n) p[t] = 0.0f;
}

// f32 -> bf16 features + squared norms from the bf16-rounded values
// (self-consistent with the bf16 Gram matrix used for ranking).
__global__ void prep_x_kernel(const float* __restrict__ x, __bf16* __restrict__ xb,
                              float* __restrict__ sq) {
  const int n = blockIdx.x * 256 + threadIdx.x;
  if (n >= NPTS) return;
  float s = 0.0f;
#pragma unroll
  for (int c = 0; c < DIN1; ++c) {
    const __bf16 vb = (__bf16)x[n * DIN1 + c];
    xb[n * DIN1 + c] = vb;
    const float vf = (float)vb;
    s += vf * vf;
  }
  sq[n] = s;
}

// W [O][K] f32 -> Wt [K][O] bf16
__global__ void wtrans_kernel(const float* __restrict__ W, __bf16* __restrict__ Wt,
                              int O, int Kd) {
  const int t = blockIdx.x * 256 + threadIdx.x;
  if (t >= O * Kd) return;
  const int o = t / Kd, k = t % Kd;
  Wt[k * O + o] = (__bf16)W[o * Kd + k];
}

// Apply row-BN (biased stats) + ReLU, emit bf16 features + sq norms.
__global__ void prep_h_kernel(const float* __restrict__ h, const float* __restrict__ rstats,
                              const float* __restrict__ g, const float* __restrict__ b,
                              __bf16* __restrict__ hb, float* __restrict__ sq) {
  const int n = blockIdx.x * 256 + threadIdx.x;
  if (n >= NPTS) return;
  const float cnt = (float)NPTS;
  float s = 0.0f;
  for (int c = 0; c < DOUT; ++c) {
    const float mean = rstats[c] / cnt;
    const float var  = rstats[DOUT + c] / cnt - mean * mean;
    float v = (h[n * DOUT + c] - mean) * rsqrtf(var + EPSBN) * g[c] + b[c];
    v = fmaxf(v, 0.0f);
    const __bf16 vb = (__bf16)v;
    hb[n * DOUT + c] = vb;
    const float vf = (float)vb;
    s += vf * vf;
  }
  sq[n] = s;
}

// --- kNN: WMMA Gram tiles, async double-buffered candidates, top-16 ---------
// Block = 256 threads (8 waves) handles 64 query rows of one graph.
// Candidate tiles (CT cols) are staged LDS-direct with
// global_load_async_to_lds_b128 and double-buffered so the next tile streams
// in while WMMA consumes the current one. d2 = |xi|^2+|xj|^2-2*xi.xj goes to
// an LDS tile; 64 scan threads keep a sorted top-16 (smallest d2, self incl).
template <int D>
__global__ __launch_bounds__(256) void knn_kernel(const __bf16* __restrict__ X,
                                                  const float* __restrict__ sq,
                                                  int* __restrict__ idx_out) {
  constexpr int CT = (D == 64) ? 64 : 32;   // candidate tile columns
  constexpr int NB = CT / 16;               // 16-wide col blocks per tile
  constexpr int ABYTES = 64 * D * 2;        // query tile bytes
  constexpr int CBYTES = CT * D * 2;        // candidate tile bytes
  constexpr int NASY   = CBYTES / 4096;     // ASYNCcnt ops per candidate copy
  constexpr int NT     = NPER / CT;

  __shared__ __bf16 Aq[64 * D];
  __shared__ __bf16 Cc[2][CT * D];
  __shared__ float  dist[64 * CT];

  const int g    = blockIdx.x >> 6;         // 64 row-tiles per graph
  const int rt   = blockIdx.x & 63;
  const int row0 = g * NPER + rt * 64;
  const int tid  = threadIdx.x;
  const int wave = tid >> 5, lane = tid & 31;
  const int rb = wave & 3, cgrp = wave >> 2;

  // Async-stage query tile + first candidate tile (LDS-direct).
  async_tile_copy<ABYTES>(Aq, X + (size_t)row0 * D, tid);
  async_tile_copy<CBYTES>(Cc[0], X + (size_t)g * NPER * D, tid);

  float best[KNN];
  int   bidx[KNN];
#pragma unroll
  for (int k = 0; k < KNN; ++k) { best[k] = 3.0e38f; bidx[k] = row0; }

  for (int ct = 0; ct < NT; ++ct) {
    const int buf  = ct & 1;
    const int col0 = g * NPER + ct * CT;
    if (ct + 1 < NT) {  // prefetch next candidate tile into the other buffer
      async_tile_copy<CBYTES>(Cc[buf ^ 1], X + (size_t)(col0 + CT) * D, tid);
      wait_async<NASY>();   // in-order: current tile (older ops) now resident
    } else {
      wait_async<0>();
    }
    __syncthreads();        // copies visible; prior scan done before dist reuse

#pragma unroll
    for (int cb = cgrp; cb < NB; cb += 2) {
      v8f acc = {};
#pragma unroll
      for (int kk = 0; kk < D; kk += 32) {
        v16bf a = lds_A_16x32(&Aq[(rb * 16) * D + kk], D, lane);
        v16bf b = lds_BT_16x32(&Cc[buf][(cb * 16) * D + kk], D, lane);
        acc = __builtin_amdgcn_wmma_f32_16x16x32_bf16(false, a, false, b,
                                                      (short)0, acc, false, false);
      }
      const int   n   = lane & 15, half = lane >> 4;
      const float sqc = sq[col0 + cb * 16 + n];
#pragma unroll
      for (int r = 0; r < 8; ++r) {
        const int m = rb * 16 + r + 8 * half;  // D layout: M = r + 8*half
        dist[m * CT + cb * 16 + n] = sq[row0 + m] + sqc - 2.0f * acc[r];
      }
    }
    __syncthreads();
    if (tid < 64) {
      for (int c = 0; c < CT; ++c) {
        const float dv = dist[tid * CT + c];
        if (dv < best[KNN - 1]) {
          best[KNN - 1] = dv;
          bidx[KNN - 1] = col0 + c;
#pragma unroll
          for (int p = KNN - 1; p > 0; --p) {  // one bubble pass re-sorts
            if (best[p] < best[p - 1]) {
              const float tf = best[p]; best[p] = best[p - 1]; best[p - 1] = tf;
              const int   ti = bidx[p]; bidx[p] = bidx[p - 1]; bidx[p - 1] = ti;
            }
          }
        }
      }
    }
  }
  if (tid < 64) {
#pragma unroll
    for (int k = 0; k < KNN; ++k) idx_out[(row0 + tid) * KNN + k] = bidx[k];
  }
}

// --- EdgeConv: WMMA edge-MLP, two passes ------------------------------------
// Block = 256 threads, tile = 4 points = 64 edges (16-edge WMMA row-block ==
// one point's k=16 neighborhood). msg = [xi, xj-xi] staged bf16 in LDS;
// weights read K-major from L2-resident Wt (prefetched).
// PASS2=false: accumulate per-channel sum/sumsq of pre-BN h (edge BN stats).
// PASS2=true : edge BN + LeakyReLU(0.2) + max over k, write h_out, row stats.
template <int DIN, bool PASS2>
__global__ __launch_bounds__(256) void edgeconv_kernel(
    const __bf16* __restrict__ X, const int* __restrict__ idx,
    const __bf16* __restrict__ Wt,
    const float* __restrict__ estats, const float* __restrict__ egamma,
    const float* __restrict__ ebeta,
    float* __restrict__ h_out, float* __restrict__ stats_out) {
  constexpr int CD = 2 * DIN;
  __shared__ __bf16 msg[64 * CD];
  __shared__ float  lstat[2 * DOUT];
  const int tid  = threadIdx.x;
  const int wave = tid >> 5, lane = tid & 31;
  const int n0   = blockIdx.x * 4;

  // Warm weights into cache while we gather the message tile.
  if (tid * 256 < CD * DOUT * 2)
    __builtin_prefetch((const char*)Wt + tid * 256, 0, 3);

  for (int t = tid; t < 2 * DOUT; t += 256) lstat[t] = 0.0f;
  for (int t = tid; t < 64 * CD; t += 256) {
    const int e = t / CD, c = t % CD;
    const int n = n0 + (e >> 4);
    const int j = idx[n * KNN + (e & 15)];
    __bf16 v;
    if (c < DIN) {
      v = X[n * DIN + c];
    } else {
      const int cc = c - DIN;
      v = (__bf16)((float)X[j * DIN + cc] - (float)X[n * DIN + cc]);
    }
    msg[t] = v;
  }
  __syncthreads();

  const int rb    = wave & 3;          // point n0+rb
  const int cbase = (wave >> 2) * 4;   // channel half
  const int nIdx  = lane & 15, half = lane >> 4;
#pragma unroll
  for (int ci = 0; ci < 4; ++ci) {
    const int cb = cbase + ci;
    v8f acc = {};
#pragma unroll
    for (int kk = 0; kk < CD; kk += 32) {
      v16bf a = lds_A_16x32(&msg[(rb * 16) * CD + kk], CD, lane);
      v16bf b = gbl_B_Kmajor(Wt, kk, cb * 16, lane);
      acc = __builtin_amdgcn_wmma_f32_16x16x32_bf16(false, a, false, b,
                                                    (short)0, acc, false, false);
    }
    const int ch = cb * 16 + nIdx;
    if (!PASS2) {
      float s = 0.0f, s2 = 0.0f;
#pragma unroll
      for (int r = 0; r < 8; ++r) { s += acc[r]; s2 += acc[r] * acc[r]; }
      atomicAdd(&lstat[ch], s);
      atomicAdd(&lstat[DOUT + ch], s2);
    } else {
      const float cnt  = (float)NEDGE;
      const float mean = estats[ch] / cnt;
      const float var  = estats[DOUT + ch] / cnt - mean * mean;
      const float scl  = rsqrtf(var + EPSBN) * egamma[ch];
      const float shf  = ebeta[ch];
      float mx = -3.0e38f;
#pragma unroll
      for (int r = 0; r < 8; ++r) {
        float hv = (acc[r] - mean) * scl + shf;
        hv = (hv >= 0.0f) ? hv : 0.2f * hv;   // LeakyReLU(0.2)
        mx = fmaxf(mx, hv);
      }
      mx = fmaxf(mx, __shfl_xor(mx, 16, 32)); // combine M halves
      if (half == 0) {
        h_out[(n0 + rb) * DOUT + ch] = mx;
        atomicAdd(&lstat[ch], mx);            // stats for the following row-BN
        atomicAdd(&lstat[DOUT + ch], mx * mx);
      }
    }
  }
  __syncthreads();
  for (int t = tid; t < 2 * DOUT; t += 256) atomicAdd(&stats_out[t], lstat[t]);
}

// --- projection shortcut (bug-compatible reshape) ---------------------------
// P[o][n] = sum_c Wp[o][c] * x.flat[c*N + n] + bp[o]; BN stats per o over n.
__global__ __launch_bounds__(256) void proj_kernel(const float* __restrict__ x,
                                                   const float* __restrict__ Wp,
                                                   const float* __restrict__ bp,
                                                   float* __restrict__ P,
                                                   float* __restrict__ pstats) {
  __shared__ float red[2];
  const int o = blockIdx.y;
  const int n = blockIdx.x * 256 + threadIdx.x;
  if (threadIdx.x == 0) { red[0] = 0.0f; red[1] = 0.0f; }
  __syncthreads();
  float acc = bp[o];
#pragma unroll
  for (int c = 0; c < DIN1; ++c) acc += Wp[o * DIN1 + c] * x[c * NPTS + n];
  P[o * NPTS + n] = acc;
  atomicAdd(&red[0], acc);
  atomicAdd(&red[1], acc * acc);
  __syncthreads();
  if (threadIdx.x == 0) {
    atomicAdd(&pstats[o], red[0]);
    atomicAdd(&pstats[DOUT + o], red[1]);
  }
}

// --- fused epilogue: bn2(h2) + bn_p(P) (flat-reshape aliased) + ReLU --------
__global__ void final_kernel(const float* __restrict__ h2, const float* __restrict__ P,
                             const float* __restrict__ bstats, const float* __restrict__ g2,
                             const float* __restrict__ b2, const float* __restrict__ pstats,
                             const float* __restrict__ gp, const float* __restrict__ bpb,
                             float* __restrict__ out) {
  const int t = blockIdx.x * 256 + threadIdx.x;
  if (t >= NPTS * DOUT) return;
  const int j = t & (DOUT - 1);   // h2 is [n][128] row-major
  const int o = t >> 15;          // P flat is [o][N] o-major: same flat index
  const float cnt = (float)NPTS;
  const float m2 = bstats[j] / cnt;
  const float v2 = bstats[DOUT + j] / cnt - m2 * m2;
  const float hv = (h2[t] - m2) * rsqrtf(v2 + EPSBN) * g2[j] + b2[j];
  const float mp = pstats[o] / cnt;
  const float vp = pstats[DOUT + o] / cnt - mp * mp;
  const float pv = (P[t] - mp) * rsqrtf(vp + EPSBN) * gp[o] + bpb[o];
  out[t] = fmaxf(hv + pv, 0.0f);
}

// ---------------------------------------------------------------------------

extern "C" void kernel_launch(void* const* d_in, const int* in_sizes, int n_in,
                              void* d_out, int out_size, void* d_ws, size_t ws_size,
                              hipStream_t stream) {
  (void)in_sizes; (void)n_in; (void)out_size; (void)ws_size;
  const float* x   = (const float*)d_in[0];
  /* d_in[1] = batch: contiguous equal-size graphs -> unused */
  const float* W1  = (const float*)d_in[2];
  const float* g1e = (const float*)d_in[3];
  const float* b1e = (const float*)d_in[4];
  const float* W2  = (const float*)d_in[5];
  const float* g2e = (const float*)d_in[6];
  const float* b2e = (const float*)d_in[7];
  const float* g1  = (const float*)d_in[8];
  const float* b1  = (const float*)d_in[9];
  const float* g2  = (const float*)d_in[10];
  const float* b2  = (const float*)d_in[11];
  const float* Wp  = (const float*)d_in[12];
  const float* bp  = (const float*)d_in[13];
  const float* gp  = (const float*)d_in[14];
  const float* bpb = (const float*)d_in[15];
  float* out = (float*)d_out;

  char*  ws  = (char*)d_ws;
  size_t off = 0;
  auto alloc = [&](size_t bytes) -> void* {
    void* p = ws + off;
    off = (off + bytes + 255) & ~(size_t)255;
    return p;
  };
  __bf16* xbf  = (__bf16*)alloc((size_t)NPTS * DIN1 * 2);      //  4 MB
  __bf16* w1t  = (__bf16*)alloc((size_t)2 * DIN1 * DOUT * 2);  // 32 KB
  __bf16* w2t  = (__bf16*)alloc((size_t)2 * DOUT * DOUT * 2);  // 64 KB
  float*  sq1  = (float*)alloc((size_t)NPTS * 4);
  float*  sq2  = (float*)alloc((size_t)NPTS * 4);
  int*    idx1 = (int*)alloc((size_t)NPTS * KNN * 4);          //  2 MB
  int*    idx2 = (int*)alloc((size_t)NPTS * KNN * 4);
  float*  h1   = (float*)alloc((size_t)NPTS * DOUT * 4);       // 16 MB
  __bf16* h1b  = (__bf16*)alloc((size_t)NPTS * DOUT * 2);      //  8 MB
  float*  h2   = (float*)alloc((size_t)NPTS * DOUT * 4);       // 16 MB
  float*  P    = (float*)alloc((size_t)NPTS * DOUT * 4);       // 16 MB
  float*  st   = (float*)alloc((size_t)5 * 2 * DOUT * 4);      // stats accumulators
  float* s_e1 = st;            // edgeconv1 edge-BN  [sum|sumsq]
  float* s_b1 = st + 256;      // bn1 row stats
  float* s_e2 = st + 512;      // edgeconv2 edge-BN
  float* s_b2 = st + 768;      // bn2 row stats
  float* s_p  = st + 1024;     // projection BN

  zero_kernel<<<5, 256, 0, stream>>>(st, 5 * 2 * DOUT);
  prep_x_kernel<<<NPTS / 256, 256, 0, stream>>>(x, xbf, sq1);
  wtrans_kernel<<<(DOUT * 2 * DIN1 + 255) / 256, 256, 0, stream>>>(W1, w1t, DOUT, 2 * DIN1);
  wtrans_kernel<<<(DOUT * 2 * DOUT + 255) / 256, 256, 0, stream>>>(W2, w2t, DOUT, 2 * DOUT);

  knn_kernel<DIN1><<<BGRAPHS * 64, 256, 0, stream>>>(xbf, sq1, idx1);
  edgeconv_kernel<DIN1, false><<<NPTS / 4, 256, 0, stream>>>(
      xbf, idx1, w1t, nullptr, nullptr, nullptr, nullptr, s_e1);
  edgeconv_kernel<DIN1, true><<<NPTS / 4, 256, 0, stream>>>(
      xbf, idx1, w1t, s_e1, g1e, b1e, h1, s_b1);

  prep_h_kernel<<<NPTS / 256, 256, 0, stream>>>(h1, s_b1, g1, b1, h1b, sq2);

  knn_kernel<DOUT><<<BGRAPHS * 64, 256, 0, stream>>>(h1b, sq2, idx2);
  edgeconv_kernel<DOUT, false><<<NPTS / 4, 256, 0, stream>>>(
      h1b, idx2, w2t, nullptr, nullptr, nullptr, nullptr, s_e2);
  edgeconv_kernel<DOUT, true><<<NPTS / 4, 256, 0, stream>>>(
      h1b, idx2, w2t, s_e2, g2e, b2e, h2, s_b2);

  proj_kernel<<<dim3(NPTS / 256, DOUT), 256, 0, stream>>>(x, Wp, bp, P, s_p);
  final_kernel<<<(NPTS * DOUT) / 256, 256, 0, stream>>>(h2, P, s_b2, g2, b2,
                                                        s_p, gp, bpb, out);
}
